// Bottleneck_43576738185626
// MI455X (gfx1250) — compile-verified, matrix operand order checked by reading, not run on hardware
//
#include <hip/hip_runtime.h>

typedef __attribute__((ext_vector_type(16))) __bf16        v16bf;
typedef __attribute__((ext_vector_type(8)))  float         v8f;
typedef __attribute__((ext_vector_type(8)))  unsigned int  v8u;

#define EPSV 1e-5f

// ---------------- workspace layout (byte offsets, 256B aligned) ----------------
#define WS_POOLED   0u            // 8*1024*64 f32            = 2,097,152
#define WS_VEC1     2097152u      // 8192 f32
#define WS_VEC2     2129920u      // 2048 f32
#define WS_MASKS    2138112u      // 512 f32 (8*64)
#define WS_MC1      2140160u      // 2048 f32
#define WS_MC2      2148352u      // 2048 f32
#define WS_NORMS    2156544u      // 8 f32
#define WS_NORMC1   2156800u      // 8 f32
#define WS_NORMC2   2157056u      // 8 f32
#define WS_SC1      2157312u      // 256 f32
#define WS_SH1      (WS_SC1 + 1024u)
#define WS_SC2      (WS_SH1 + 1024u)
#define WS_SH2      (WS_SC2 + 1024u)
#define WS_SC3      (WS_SH2 + 1024u)   // 1024 f32
#define WS_SH3      (WS_SC3 + 4096u)   // 1024 f32
#define WS_W1B      2169600u      // 262144 bf16 = 524,288 B
#define WS_W2B      2693888u      // 589824 bf16 = 1,179,648 B  ([tap][co][ci])
#define WS_W3B      3873536u      // 262144 bf16 = 524,288 B
#define WS_OUT1B    4397824u      // 8*256*4096 bf16 = 16 MB
#define WS_OUT2B    21175040u     // 8*256*4096 bf16 = 16 MB
// total ~ 38 MB

// ---------------- CDNA5 async global->LDS copy (32 bytes / lane) ----------------
__device__ __forceinline__ void async_cp16x2(const void* gbase, unsigned goff, void* ldsdst) {
  unsigned l = (unsigned)(size_t)ldsdst;                 // low 32 bits = LDS byte offset
  unsigned long long b = (unsigned long long)(size_t)gbase;
  asm volatile("global_load_async_to_lds_b128 %0, %1, %2"
               :: "v"(l), "v"(goff), "s"(b) : "memory");
  asm volatile("global_load_async_to_lds_b128 %0, %1, %2 offset:16"
               :: "v"(l), "v"(goff), "s"(b) : "memory");
}
#define WAIT_ASYNC() asm volatile("s_wait_asynccnt 0" ::: "memory")

#define WMMA_BF16(A, B, C) \
  __builtin_amdgcn_wmma_f32_16x16x32_bf16(false, (A), false, (B), (short)0, (C), false, false)

// Fragment fill per CDNA5 16-bit A/B layout (lane 0-15: K pairs {0..7}|{16..23},
// lane 16-31: K pairs {8..15}|{24..31}); LDS rows are 32 bf16 = 16 dwords.
// The two 4-dword runs per fragment vectorize into ds_load_b128 pairs.
__device__ __forceinline__ v16bf frag_ld(const unsigned* base32, int rowBase, int lane) {
  const int half = (lane >> 4) & 1, r = lane & 15;
  const unsigned* row = base32 + (size_t)(rowBase + r) * 16;
  v8u u;
#pragma unroll
  for (int p = 0; p < 8; ++p)
    u[p] = row[(p >> 2) * 8 + half * 4 + (p & 3)];
  return __builtin_bit_cast(v16bf, u);
}

__device__ __forceinline__ void mma_stage(const __bf16* As, const __bf16* Bs,
                                          int wm, int wn, int lane, v8f acc[4][2]) {
  const unsigned* A32 = (const unsigned*)As;
  const unsigned* B32 = (const unsigned*)Bs;
  v16bf a[4], b[2];
#pragma unroll
  for (int i = 0; i < 4; ++i) a[i] = frag_ld(A32, wm * 64 + i * 16, lane);
#pragma unroll
  for (int j = 0; j < 2; ++j) b[j] = frag_ld(B32, wn * 32 + j * 16, lane);
#pragma unroll
  for (int i = 0; i < 4; ++i)
#pragma unroll
    for (int j = 0; j < 2; ++j)
      acc[i][j] = WMMA_BF16(a[i], b[j], acc[i][j]);
}

// ================= stage kernels =================

// pooled 8x8 block means + GAP per (n,c). grid 8192, block 256.
__global__ __launch_bounds__(256) void k_stats(const float* __restrict__ x,
                                               float* __restrict__ pooled,
                                               float* __restrict__ vec1) {
  const int nc = blockIdx.x;
  const float* px = x + ((size_t)nc << 12);
  const int t = threadIdx.x, h = t >> 2, sg = t & 3;
  const float4* p4 = (const float4*)(px + h * 64 + sg * 16);
  float4 f0 = p4[0], f1 = p4[1], f2 = p4[2], f3 = p4[3];
  const float s0 = f0.x + f0.y + f0.z + f0.w + f1.x + f1.y + f1.z + f1.w;
  const float s1 = f2.x + f2.y + f2.z + f2.w + f3.x + f3.y + f3.z + f3.w;
  __shared__ float part[512];
  __shared__ float bsum[64];
  part[h * 8 + sg * 2]     = s0;
  part[h * 8 + sg * 2 + 1] = s1;
  __syncthreads();
  if (t < 64) {
    const int bh = t >> 3, bw = t & 7;
    float s = 0.f;
#pragma unroll
    for (int r = 0; r < 8; ++r) s += part[(bh * 8 + r) * 8 + bw];
    bsum[t] = s;
    pooled[(size_t)nc * 64 + t] = s * (1.f / 64.f);
  }
  __syncthreads();
  if (t == 0) {
    float tot = 0.f;
    for (int i = 0; i < 64; ++i) tot += bsum[i];
    vec1[nc] = tot * (1.f / 4096.f);
  }
}

// 3x3 conv 1024->1 on pooled (pad 1), hard gate. grid 8, block 256.
__global__ __launch_bounds__(256) void k_mask_s(const float* __restrict__ pooled,
                                                const float* __restrict__ wsc,
                                                const float* __restrict__ wsb,
                                                float* __restrict__ mask_s,
                                                float* __restrict__ norm_s) {
  const int n = blockIdx.x, t = threadIdx.x;
  const int pix = t >> 2, oh = pix >> 3, ow = pix & 7, cpart = t & 3;
  const float* pc = pooled + (size_t)(n * 1024 + cpart * 256) * 64;
  const float* wp = wsc + (size_t)(cpart * 256) * 9;
  float s = 0.f;
  for (int dy = -1; dy <= 1; ++dy)
    for (int dx = -1; dx <= 1; ++dx) {
      const int ih = oh + dy, iw = ow + dx;
      if ((unsigned)ih < 8u && (unsigned)iw < 8u) {
        const int off = ih * 8 + iw;
        const int wtap = (dy + 1) * 3 + (dx + 1);
        for (int c = 0; c < 256; ++c)
          s += pc[c * 64 + off] * wp[c * 9 + wtap];
      }
    }
  __shared__ float red[256];
  __shared__ float mk[64];
  red[t] = s;
  __syncthreads();
  if (cpart == 0) {
    const float logit = red[t] + red[t + 1] + red[t + 2] + red[t + 3] + wsb[0];
    const float m = (logit >= 0.f) ? 1.f : 0.f;
    mask_s[n * 64 + pix] = m;
    mk[pix] = m;
  }
  __syncthreads();
  if (t == 0) {
    float tot = 0.f;
    for (int i = 0; i < 64; ++i) tot += mk[i];
    norm_s[n] = tot;
  }
}

// FC -> hard gate + count. grid 8, block 256 (one thread per output channel).
__global__ __launch_bounds__(256) void k_fcmask(const float* __restrict__ vec,
                                                const float* __restrict__ wfc,
                                                const float* __restrict__ bfc, int K,
                                                float* __restrict__ mcout,
                                                float* __restrict__ normout) {
  const int n = blockIdx.x, p = threadIdx.x;
  const float* v = vec + (size_t)n * K;
  const float* wr = wfc + (size_t)p * K;
  float s = bfc[p];
  for (int k = 0; k < K; k += 4)
    s += v[k] * wr[k] + v[k + 1] * wr[k + 1] + v[k + 2] * wr[k + 2] + v[k + 3] * wr[k + 3];
  const float m = (s >= 0.f) ? 1.f : 0.f;
  mcout[n * 256 + p] = m;
  __shared__ float red[256];
  red[p] = m;
  __syncthreads();
  for (int off = 128; off > 0; off >>= 1) {
    if (p < off) red[p] += red[p + off];
    __syncthreads();
  }
  if (p == 0) normout[n] = red[0];
}

// fold BN into scale/shift. grid 6, block 256.
__global__ void k_bnprep(const float* __restrict__ bn1, const float* __restrict__ bn2,
                         const float* __restrict__ bn3,
                         float* sc1, float* sh1, float* sc2, float* sh2,
                         float* sc3, float* sh3) {
  const int i = blockIdx.x * 256 + threadIdx.x;
  if (i < 256) {
    const int c = i, C = 256;
    const float s = bn1[c] * rsqrtf(bn1[3 * C + c] + EPSV);
    sc1[c] = s; sh1[c] = bn1[C + c] - bn1[2 * C + c] * s;
  } else if (i < 512) {
    const int c = i - 256, C = 256;
    const float s = bn2[c] * rsqrtf(bn2[3 * C + c] + EPSV);
    sc2[c] = s; sh2[c] = bn2[C + c] - bn2[2 * C + c] * s;
  } else if (i < 1536) {
    const int c = i - 512, C = 1024;
    const float s = bn3[c] * rsqrtf(bn3[3 * C + c] + EPSV);
    sc3[c] = s; sh3[c] = bn3[C + c] - bn3[2 * C + c] * s;
  }
}

// weights -> bf16 (w2 repacked to [tap][co][ci]). grid 4352, block 256.
__global__ void k_wprep(const float* __restrict__ w1, const float* __restrict__ w2,
                        const float* __restrict__ w3,
                        __bf16* __restrict__ w1b, __bf16* __restrict__ w2b,
                        __bf16* __restrict__ w3b) {
  const int idx = blockIdx.x * 256 + threadIdx.x;
  if (idx < 262144) {
    w1b[idx] = (__bf16)w1[idx];
  } else if (idx < 262144 + 589824) {
    const int rel = idx - 262144;
    const int tap = rel >> 16, r2 = rel & 65535;
    const int co = r2 >> 8, ci = r2 & 255;
    w2b[rel] = (__bf16)w2[(co * 256 + ci) * 9 + tap];
  } else if (idx < 1114112) {
    const int rel = idx - 262144 - 589824;
    w3b[rel] = (__bf16)w3[rel];
  }
}

// conv1x1 1024->256 as WMMA GEMM + BN + ReLU + mc1*mask_s gate. grid (256,2), block 256.
// Ping-pong LDS double buffering: async DMA of next tile overlaps WMMA on current.
__global__ __launch_bounds__(256) void k_gemm_conv1(
    const float* __restrict__ x, const __bf16* __restrict__ w1b,
    const float* __restrict__ sc, const float* __restrict__ sh,
    const float* __restrict__ mc1, const float* __restrict__ mask_s,
    __bf16* __restrict__ out1b) {
  __shared__ __bf16 As[2][128 * 32];   // [co][k]
  __shared__ __bf16 Bs[2][128 * 32];   // [col][k]
  const int t = threadIdx.x;
  const int j0 = blockIdx.x * 128;
  const int n = j0 >> 12, hw0 = j0 & 4095;
  const int m0 = blockIdx.y * 128;
  const int wave = t >> 5, lane = t & 31;
  const int wm = wave & 1, wn = wave >> 1;
  const int arow = t >> 1, aseg = t & 1;
  const int kb = t >> 3, segb = t & 7;
  v8f acc[4][2] = {};

  auto load_tiles = [&](int kc, int buf) {
    async_cp16x2(w1b, (unsigned)(((m0 + arow) * 1024 + kc + aseg * 16) * 2),
                 &As[buf][arow * 32 + aseg * 16]);
    const float* gB = x + (((size_t)(n * 1024 + kc + kb)) << 12) + hw0 + segb * 16;
    float4 ff[4];
    ff[0] = ((const float4*)gB)[0]; ff[1] = ((const float4*)gB)[1];
    ff[2] = ((const float4*)gB)[2]; ff[3] = ((const float4*)gB)[3];
    const float* fv = (const float*)ff;
#pragma unroll
    for (int i = 0; i < 16; ++i)
      Bs[buf][(segb * 16 + i) * 32 + kb] = (__bf16)fv[i];
  };

  load_tiles(0, 0);
  WAIT_ASYNC();
  __syncthreads();
  int buf = 0;
  for (int kc = 0; kc < 1024; kc += 32) {
    if (kc + 32 < 1024) load_tiles(kc + 32, buf ^ 1);   // DMA overlaps WMMA below
    mma_stage(As[buf], Bs[buf], wm, wn, lane, acc);
    WAIT_ASYNC();
    __syncthreads();
    buf ^= 1;
  }
#pragma unroll
  for (int j = 0; j < 2; ++j) {
    const int col = wn * 32 + j * 16 + (lane & 15);
    const int hw = hw0 + col, h = hw >> 6, w = hw & 63;
    const float msv = mask_s[n * 64 + (h >> 3) * 8 + (w >> 3)];
#pragma unroll
    for (int i = 0; i < 4; ++i) {
#pragma unroll
      for (int r = 0; r < 8; ++r) {
        const int co = m0 + wm * 64 + i * 16 + (lane >> 4) * 8 + r;
        float v = acc[i][j][r] * sc[co] + sh[co];
        v = fmaxf(v, 0.f);
        v *= mc1[n * 256 + co] * msv;
        out1b[(((size_t)(n * 256 + co)) << 12) + hw] = (__bf16)v;
      }
    }
  }
}

// GAP of gated out1. grid 2048, block 256.
__global__ __launch_bounds__(256) void k_vec2(const __bf16* __restrict__ out1b,
                                              float* __restrict__ vec2) {
  const int row = blockIdx.x;
  const __bf16* p = out1b + ((size_t)row << 12);
  float s = 0.f;
  for (int i = threadIdx.x; i < 4096; i += 256) s += (float)p[i];
  __shared__ float red[256];
  red[threadIdx.x] = s;
  __syncthreads();
  for (int off = 128; off > 0; off >>= 1) {
    if (threadIdx.x < off) red[threadIdx.x] += red[threadIdx.x + off];
    __syncthreads();
  }
  if (threadIdx.x == 0) vec2[row] = red[0] * (1.f / 4096.f);
}

// conv3x3 256->256 as 9 tap-accumulated WMMA GEMMs + BN + ReLU + mc2*mask_s.
// 72 pipelined steps (tap-major) with ping-pong LDS.
__global__ __launch_bounds__(256) void k_gemm_conv2(
    const __bf16* __restrict__ out1b, const __bf16* __restrict__ w2b,
    const float* __restrict__ sc, const float* __restrict__ sh,
    const float* __restrict__ mc2, const float* __restrict__ mask_s,
    __bf16* __restrict__ out2b) {
  __shared__ __bf16 As[2][128 * 32];
  __shared__ __bf16 Bs[2][128 * 32];
  const int t = threadIdx.x;
  const int j0 = blockIdx.x * 128;
  const int n = j0 >> 12, hw0 = j0 & 4095;
  const int m0 = blockIdx.y * 128;
  const int wave = t >> 5, lane = t & 31;
  const int wm = wave & 1, wn = wave >> 1;
  const int arow = t >> 1, aseg = t & 1;
  const int kb = t >> 3, segb = t & 7;
  v8f acc[4][2] = {};

  auto load_tiles = [&](int s, int buf) {
    const int tap = s >> 3, kc = (s & 7) * 32;
    const int dy = tap / 3 - 1, dx = tap % 3 - 1;
    async_cp16x2(w2b,
                 (unsigned)((tap * 65536 + (m0 + arow) * 256 + kc + aseg * 16) * 2),
                 &As[buf][arow * 32 + aseg * 16]);
    const __bf16* src = out1b + (((size_t)(n * 256 + kc + kb)) << 12);
    __builtin_prefetch((const void*)(src + 4096), 0, 1);
#pragma unroll
    for (int i = 0; i < 16; ++i) {
      const int hw = hw0 + segb * 16 + i;
      const int h = hw >> 6, w = hw & 63;
      const int hs = h + dy, ws2 = w + dx;
      __bf16 v = (__bf16)0.f;
      if ((unsigned)hs < 64u && (unsigned)ws2 < 64u) v = src[hs * 64 + ws2];
      Bs[buf][(segb * 16 + i) * 32 + kb] = v;
    }
  };

  load_tiles(0, 0);
  WAIT_ASYNC();
  __syncthreads();
  int buf = 0;
  for (int s = 0; s < 72; ++s) {
    if (s + 1 < 72) load_tiles(s + 1, buf ^ 1);
    mma_stage(As[buf], Bs[buf], wm, wn, lane, acc);
    WAIT_ASYNC();
    __syncthreads();
    buf ^= 1;
  }
#pragma unroll
  for (int j = 0; j < 2; ++j) {
    const int col = wn * 32 + j * 16 + (lane & 15);
    const int hw = hw0 + col, h = hw >> 6, w = hw & 63;
    const float msv = mask_s[n * 64 + (h >> 3) * 8 + (w >> 3)];
#pragma unroll
    for (int i = 0; i < 4; ++i) {
#pragma unroll
      for (int r = 0; r < 8; ++r) {
        const int co = m0 + wm * 64 + i * 16 + (lane >> 4) * 8 + r;
        float v = acc[i][j][r] * sc[co] + sh[co];
        v = fmaxf(v, 0.f);
        v *= mc2[n * 256 + co] * msv;
        out2b[(((size_t)(n * 256 + co)) << 12) + hw] = (__bf16)v;
      }
    }
  }
}

// conv1x1 256->1024 + BN + mask_s + residual + ReLU -> fp32 out. grid (256,8).
__global__ __launch_bounds__(256) void k_gemm_conv3(
    const __bf16* __restrict__ out2b, const __bf16* __restrict__ w3b,
    const float* __restrict__ sc, const float* __restrict__ sh,
    const float* __restrict__ mask_s, const float* __restrict__ x,
    float* __restrict__ outp) {
  __shared__ __bf16 As[2][128 * 32];
  __shared__ __bf16 Bs[2][128 * 32];
  const int t = threadIdx.x;
  const int j0 = blockIdx.x * 128;
  const int n = j0 >> 12, hw0 = j0 & 4095;
  const int m0 = blockIdx.y * 128;
  const int wave = t >> 5, lane = t & 31;
  const int wm = wave & 1, wn = wave >> 1;
  const int arow = t >> 1, aseg = t & 1;
  const int kb = t >> 3, segb = t & 7;
  v8f acc[4][2] = {};

  auto load_tiles = [&](int kc, int buf) {
    async_cp16x2(w3b, (unsigned)(((m0 + arow) * 256 + kc + aseg * 16) * 2),
                 &As[buf][arow * 32 + aseg * 16]);
    const __bf16* src = out2b + (((size_t)(n * 256 + kc + kb)) << 12) + hw0 + segb * 16;
    uint4 uu[2];
    uu[0] = ((const uint4*)src)[0];
    uu[1] = ((const uint4*)src)[1];
    const __bf16* tmp = (const __bf16*)uu;
#pragma unroll
    for (int i = 0; i < 16; ++i)
      Bs[buf][(segb * 16 + i) * 32 + kb] = tmp[i];
  };

  load_tiles(0, 0);
  WAIT_ASYNC();
  __syncthreads();
  int buf = 0;
  for (int kc = 0; kc < 256; kc += 32) {
    if (kc + 32 < 256) load_tiles(kc + 32, buf ^ 1);
    mma_stage(As[buf], Bs[buf], wm, wn, lane, acc);
    WAIT_ASYNC();
    __syncthreads();
    buf ^= 1;
  }
#pragma unroll
  for (int j = 0; j < 2; ++j) {
    const int col = wn * 32 + j * 16 + (lane & 15);
    const int hw = hw0 + col, h = hw >> 6, w = hw & 63;
    const float msv = mask_s[n * 64 + (h >> 3) * 8 + (w >> 3)];
#pragma unroll
    for (int i = 0; i < 4; ++i) {
#pragma unroll
      for (int r = 0; r < 8; ++r) {
        const int co = m0 + wm * 64 + i * 16 + (lane >> 4) * 8 + r;
        const size_t idx = (((size_t)(n * 1024 + co)) << 12) + hw;
        float v = acc[i][j][r] * sc[co] + sh[co];
        v = v * msv + x[idx];
        outp[idx] = fmaxf(v, 0.f);
      }
    }
  }
}

// bookkeeping tail of d_out. grid 1, block 32.
__global__ void k_book(const float* __restrict__ n1in, const float* __restrict__ n2in,
                       const float* __restrict__ flin, const float* __restrict__ norm_s,
                       const float* __restrict__ norm_c1, const float* __restrict__ norm_c2,
                       float* __restrict__ outp) {
  const int t = threadIdx.x;
  float* b1 = outp + 33554432;  // norm_1: 2x9
  float* b2 = b1 + 18;          // norm_2: 3x9
  float* bf = b2 + 27;          // flops:  2x9
  if (t < 9) { b1[t] = n1in[t]; b2[t] = n2in[t]; bf[t] = flin[t]; }
  if (t < 8) {
    b1[9 + t] = norm_s[t];
    b2[9 + t] = norm_c1[t];
    b2[18 + t] = norm_c2[t];
    const float ss = 64.f * norm_s[t];
    bf[9 + t] = ss * norm_c1[t] * 1024.f
              + 9.f * ss * norm_c2[t] * norm_c1[t]
              + ss * 1024.f * norm_c2[t];
  }
  if (t == 8) {
    b1[17] = 64.f; b2[17] = 256.f; b2[26] = 256.f;
    bf[17] = (float)(2.0 * 4096.0 * 256.0 * 1024.0 + 9.0 * 4096.0 * 256.0 * 256.0);
  }
}

// ================= launch =================
extern "C" void kernel_launch(void* const* d_in, const int* in_sizes, int n_in,
                              void* d_out, int out_size, void* d_ws, size_t ws_size,
                              hipStream_t stream) {
  const float* x       = (const float*)d_in[0];
  const float* n1in    = (const float*)d_in[1];
  const float* n2in    = (const float*)d_in[2];
  const float* flin    = (const float*)d_in[3];
  const float* ws_conv = (const float*)d_in[4];
  const float* ws_bias = (const float*)d_in[5];
  const float* wc1     = (const float*)d_in[6];
  const float* bc1     = (const float*)d_in[7];
  const float* wc2     = (const float*)d_in[8];
  const float* bc2     = (const float*)d_in[9];
  const float* w1      = (const float*)d_in[10];
  const float* bn1     = (const float*)d_in[11];
  const float* w2      = (const float*)d_in[12];
  const float* bn2     = (const float*)d_in[13];
  const float* w3      = (const float*)d_in[14];
  const float* bn3     = (const float*)d_in[15];
  char* ws = (char*)d_ws;
  float*  pooled  = (float*)(ws + WS_POOLED);
  float*  vec1    = (float*)(ws + WS_VEC1);
  float*  vec2    = (float*)(ws + WS_VEC2);
  float*  mask_s  = (float*)(ws + WS_MASKS);
  float*  mc1     = (float*)(ws + WS_MC1);
  float*  mc2     = (float*)(ws + WS_MC2);
  float*  norm_s  = (float*)(ws + WS_NORMS);
  float*  norm_c1 = (float*)(ws + WS_NORMC1);
  float*  norm_c2 = (float*)(ws + WS_NORMC2);
  float*  sc1 = (float*)(ws + WS_SC1), *sh1 = (float*)(ws + WS_SH1);
  float*  sc2 = (float*)(ws + WS_SC2), *sh2 = (float*)(ws + WS_SH2);
  float*  sc3 = (float*)(ws + WS_SC3), *sh3 = (float*)(ws + WS_SH3);
  __bf16* w1b   = (__bf16*)(ws + WS_W1B);
  __bf16* w2b   = (__bf16*)(ws + WS_W2B);
  __bf16* w3b   = (__bf16*)(ws + WS_W3B);
  __bf16* out1b = (__bf16*)(ws + WS_OUT1B);
  __bf16* out2b = (__bf16*)(ws + WS_OUT2B);
  float*  outp  = (float*)d_out;

  k_bnprep<<<6, 256, 0, stream>>>(bn1, bn2, bn3, sc1, sh1, sc2, sh2, sc3, sh3);
  k_wprep<<<4352, 256, 0, stream>>>(w1, w2, w3, w1b, w2b, w3b);
  k_stats<<<8192, 256, 0, stream>>>(x, pooled, vec1);
  k_mask_s<<<8, 256, 0, stream>>>(pooled, ws_conv, ws_bias, mask_s, norm_s);
  k_fcmask<<<8, 256, 0, stream>>>(vec1, wc1, bc1, 1024, mc1, norm_c1);
  k_gemm_conv1<<<dim3(256, 2), 256, 0, stream>>>(x, w1b, sc1, sh1, mc1, mask_s, out1b);
  k_vec2<<<2048, 256, 0, stream>>>(out1b, vec2);
  k_fcmask<<<8, 256, 0, stream>>>(vec2, wc2, bc2, 256, mc2, norm_c2);
  k_gemm_conv2<<<dim3(256, 2), 256, 0, stream>>>(out1b, w2b, sc2, sh2, mc2, mask_s, out2b);
  k_gemm_conv3<<<dim3(256, 8), 256, 0, stream>>>(out2b, w3b, sc3, sh3, mask_s, x, outp);
  k_book<<<1, 32, 0, stream>>>(n1in, n2in, flin, norm_s, norm_c1, norm_c2, outp);
}
